// GCNIILayer_86294482911939
// MI455X (gfx1250) — matrix operand order repaired and checked by reading, not compiled, not run on hardware
//
#include <hip/hip_runtime.h>
#include <hip/hip_bf16.h>

#define DD 128
#define ALPHA_C 0.1f
#define BETA_C 0.5f

typedef _Float16 v16h __attribute__((ext_vector_type(16)));
typedef _Float16 v8h  __attribute__((ext_vector_type(8)));
typedef _Float16 v4h  __attribute__((ext_vector_type(4)));
typedef float    v8f  __attribute__((ext_vector_type(8)));
typedef float    v4f  __attribute__((ext_vector_type(4)));

// ---- zero a float4-aligned buffer ----
__global__ void zero_f4(v4f* __restrict__ p, int n4) {
    int i = blockIdx.x * blockDim.x + threadIdx.x;
    if (i < n4) p[i] = (v4f){0.f, 0.f, 0.f, 0.f};
}

// ---- degree accumulation: degs[dst[e]] += 1 ----
__global__ void deg_k(const int* __restrict__ dst, float* __restrict__ degs, int E) {
    int e = blockIdx.x * blockDim.x + threadIdx.x;
    if (e < E) atomicAdd(&degs[dst[e]], 1.0f);
}

// ---- degs[n] <- max(deg,1)^-0.5 (overwrite in place) ----
__global__ void norm_k(float* __restrict__ degs, int N) {
    int n = blockIdx.x * blockDim.x + threadIdx.x;
    if (n < N) degs[n] = rsqrtf(fmaxf(degs[n], 1.0f));
}

// ---- h = features * norm[node], float4 vectorized ----
__global__ void scale_k(const v4f* __restrict__ feat, const float* __restrict__ norm,
                        v4f* __restrict__ h, int n4) {
    int i = blockIdx.x * blockDim.x + threadIdx.x;
    if (i >= n4) return;
    float nm = norm[i >> 5];              // 32 float4 chunks per node (D=128)
    h[i] = feat[i] * nm;
}

// ---- edge scatter: agg[dst] += h[src]; 32 lanes/edge, 4 floats/lane ----
__global__ void edge_k(const int* __restrict__ src, const int* __restrict__ dst,
                       const float* __restrict__ h, float* __restrict__ agg, int E) {
    long long t = (long long)blockIdx.x * blockDim.x + threadIdx.x;
    int e = (int)(t >> 5);
    if (e >= E) return;
    int c = (int)(t & 31) * 4;
    const v4f v = *(const v4f*)(h + (size_t)src[e] * DD + c);
    float* ap = agg + (size_t)dst[e] * DD + c;
    atomicAdd(ap + 0, v[0]);
    atomicAdd(ap + 1, v[1]);
    atomicAdd(ap + 2, v[2]);
    atomicAdd(ap + 3, v[3]);
}

// ---- h3 = (1-a)*(agg*norm) + a*init ; write f32 + f16 shadow ----
__global__ void combine_k(const v4f* __restrict__ agg, const float* __restrict__ norm,
                          const v4f* __restrict__ init, v4f* __restrict__ h3,
                          v4h* __restrict__ h16, int n4) {
    int i = blockIdx.x * blockDim.x + threadIdx.x;
    if (i >= n4) return;
    float nm = norm[i >> 5];
    v4f a = agg[i] * nm;
    v4f r = a * (1.0f - ALPHA_C) + init[i] * ALPHA_C;
    h3[i] = r;
    v4h hh;
    hh[0] = (_Float16)r[0]; hh[1] = (_Float16)r[1];
    hh[2] = (_Float16)r[2]; hh[3] = (_Float16)r[3];
    h16[i] = hh;
}

// ---- W -> f16 ----
__global__ void wconv_k(const float* __restrict__ W, _Float16* __restrict__ w16, int n) {
    int i = blockIdx.x * blockDim.x + threadIdx.x;
    if (i < n) w16[i] = (_Float16)W[i];
}

// ---- WMMA GEMM + mix + ReLU epilogue ----
// Each wave: 16-row stripe, all 128 cols. out = relu(0.5*h3 + 0.5*(h3 @ W^T))
__global__ void __launch_bounds__(256)
gemm_k(const _Float16* __restrict__ h16, const _Float16* __restrict__ w16,
       const float* __restrict__ h3, float* __restrict__ out, int ntiles) {
    const int wave = threadIdx.x >> 5;
    const int lane = threadIdx.x & 31;
    const int tile = blockIdx.x * 8 + wave;
    if (tile >= ntiles) return;                  // wave-uniform: EXEC stays all-ones

    const int r   = lane & 15;                   // A row / B col / C col within tile
    const int hi  = lane >> 4;
    const int akb = hi * 8;                      // A K-subchunk base (halves)
    const int bkb = hi * 8;                      // B K-subchunk base (halves)
    const int mb  = hi * 8;                      // C/D row base

    // Preload A: 16 rows x 128 K as four 16x32 f16 tiles
    const _Float16* ap = h16 + (size_t)(tile * 16 + r) * DD;
    v16h A[4];
#pragma unroll
    for (int kt = 0; kt < 4; ++kt) {
        union { v16h v; v8h h[2]; } u;
        u.h[0] = *(const v8h*)(ap + kt * 32 + akb);
        u.h[1] = *(const v8h*)(ap + kt * 32 + 16 + akb);
        A[kt] = u.v;
    }

#pragma unroll
    for (int jt = 0; jt < 8; ++jt) {
        v8f acc = {0.f, 0.f, 0.f, 0.f, 0.f, 0.f, 0.f, 0.f};
        // B element i (k = bkb+i for i<8, 16+bkb+(i-8) for i>=8) of column n = W[n][k] -> row read of W
        const _Float16* bp = w16 + (size_t)(jt * 16 + r) * DD + bkb;
#pragma unroll
        for (int kt = 0; kt < 4; ++kt) {
            union { v16h v; v8h h[2]; } bu;
            bu.h[0] = *(const v8h*)(bp + kt * 32);
            bu.h[1] = *(const v8h*)(bp + kt * 32 + 16);
            acc = __builtin_amdgcn_wmma_f32_16x16x32_f16(
                false, A[kt], false, bu.v, (short)0, acc, false, false);
        }
        // Epilogue: D layout lane(l)/vgpr(v) -> row mb+v, col r
#pragma unroll
        for (int v = 0; v < 8; ++v) {
            const size_t row = (size_t)tile * 16 + mb + v;
            const int col = jt * 16 + r;
            const float pre = h3[row * DD + col];
            float o = (1.0f - BETA_C) * pre + BETA_C * acc[v];
            out[row * DD + col] = fmaxf(o, 0.0f);
        }
    }
}

extern "C" void kernel_launch(void* const* d_in, const int* in_sizes, int n_in,
                              void* d_out, int out_size, void* d_ws, size_t ws_size,
                              hipStream_t stream) {
    const float* feat = (const float*)d_in[0];
    const float* init = (const float*)d_in[1];
    const float* W    = (const float*)d_in[2];
    const int*   src  = (const int*)d_in[3];
    const int*   dst  = (const int*)d_in[4];
    const int N = in_sizes[0] / DD;
    const int E = in_sizes[3];

    // carve workspace (256B-aligned partitions)
    char* ws = (char*)d_ws;
    size_t off = 0;
    auto carve = [&](size_t bytes) -> char* {
        char* p = ws + off;
        off += (bytes + 255) & ~(size_t)255;
        return p;
    };
    float*    degs = (float*)carve((size_t)N * 4);            // degrees -> norm
    float*    hbuf = (float*)carve((size_t)N * DD * 4);       // h, then h3 (f32)
    float*    agg  = (float*)carve((size_t)N * DD * 4);       // scatter target
    _Float16* h16  = (_Float16*)carve((size_t)N * DD * 2);    // h3 f16 shadow
    _Float16* w16  = (_Float16*)carve((size_t)DD * DD * 2);   // W f16

    const int nd4 = N * DD / 4;
    const int nn4 = N / 4;

    zero_f4<<<(nn4 + 255) / 256, 256, 0, stream>>>((v4f*)degs, nn4);
    zero_f4<<<(nd4 + 255) / 256, 256, 0, stream>>>((v4f*)agg, nd4);
    deg_k<<<(E + 255) / 256, 256, 0, stream>>>(dst, degs, E);
    norm_k<<<(N + 255) / 256, 256, 0, stream>>>(degs, N);
    scale_k<<<(nd4 + 255) / 256, 256, 0, stream>>>((const v4f*)feat, degs, (v4f*)hbuf, nd4);

    long long et = (long long)E * 32;
    edge_k<<<(unsigned)((et + 255) / 256), 256, 0, stream>>>(src, dst, hbuf, agg, E);

    combine_k<<<(nd4 + 255) / 256, 256, 0, stream>>>((const v4f*)agg, degs,
                                                     (const v4f*)init, (v4f*)hbuf,
                                                     (v4h*)h16, nd4);
    wconv_k<<<(DD * DD + 255) / 256, 256, 0, stream>>>(W, w16, DD * DD);

    const int ntiles = N / 16;   // N=100000 -> 6250 full tiles
    gemm_k<<<(ntiles + 7) / 8, 256, 0, stream>>>(h16, w16, hbuf, (float*)d_out, ntiles);
}
